// SimpleSelfAttention_25005299597452
// MI455X (gfx1250) — matrix operand order repaired and checked by reading, not compiled, not run on hardware
//
#include <hip/hip_runtime.h>
#include <math.h>

typedef __attribute__((ext_vector_type(16))) _Float16 v16h;
typedef __attribute__((ext_vector_type(8)))  _Float16 v8h;
typedef __attribute__((ext_vector_type(4)))  _Float16 v4h;
typedef __attribute__((ext_vector_type(8)))  float    v8f;

union V16U { v16h v; v8h h[2]; };

// ---------------------------------------------------------------------------
// fp32 -> fp16 cast, 4 elements per thread
// ---------------------------------------------------------------------------
__global__ __launch_bounds__(256) void cast_f32_to_f16_vec4(
    const float* __restrict__ src, _Float16* __restrict__ dst, int n4) {
  int i = blockIdx.x * blockDim.x + threadIdx.x;
  if (i < n4) {
    float4 f = ((const float4*)src)[i];
    v4h o = { (_Float16)f.x, (_Float16)f.y, (_Float16)f.z, (_Float16)f.w };
    ((v4h*)dst)[i] = o;
  }
}

// ---------------------------------------------------------------------------
// C[m,n] = sum_d A[m,d] * W[n,d] + bias[n]   (torch Linear: x @ W^T + b)
// A: MxK f16 row-major, W: NxK f16 row-major.
// transposed==0: C[m][n] f16 row-major (ld N)
// transposed==1: C stored as Ct[batch][n][s] with m = batch*Sper + s (for V^T)
// Block = 128 threads = 4 waves; block tile = 16(M) x 64(N); wave tile 16x16.
// ---------------------------------------------------------------------------
__global__ __launch_bounds__(128) void gemm_nt_bias_f16(
    const _Float16* __restrict__ A, const _Float16* __restrict__ W,
    const float* __restrict__ bias, _Float16* __restrict__ C,
    int M, int N, int K, int transposed, int Sper) {
  const int wave = threadIdx.x >> 5;
  const int lane = threadIdx.x & 31;
  const int m0 = blockIdx.x * 16;
  const int n0 = blockIdx.y * 64 + wave * 16;
  const int half8 = (lane & 16) ? 8 : 0;   // accumulator upper-lane row offset
  const int koff  = (lane & 16) ? 16 : 0;  // B-operand upper-lane K offset

  const _Float16* arow = A + (size_t)(m0 + (lane & 15)) * K;
  const _Float16* wrow = W + (size_t)(n0 + (lane & 15)) * K;

  v8f acc = {};
  for (int k0 = 0; k0 < K; k0 += 32) {
    V16U a, b;
    a.h[0] = *(const v8h*)(arow + k0 + koff);
    a.h[1] = *(const v8h*)(arow + k0 + koff + 8);
    b.h[0] = *(const v8h*)(wrow + k0 + koff);
    b.h[1] = *(const v8h*)(wrow + k0 + koff + 8);
    acc = __builtin_amdgcn_wmma_f32_16x16x32_f16(false, a.v, false, b.v,
                                                 (short)0, acc, false, false);
  }

  const int ncol = n0 + (lane & 15);
  const float bn = bias[ncol];
  if (!transposed) {
    for (int r = 0; r < 8; ++r)
      C[(size_t)(m0 + half8 + r) * N + ncol] = (_Float16)(acc[r] + bn);
  } else {
    const int m = m0 + half8;            // 8 consecutive rows m..m+7
    const int bb = m / Sper;
    const int s  = m - bb * Sper;
    v8h o;
    for (int r = 0; r < 8; ++r) o[r] = (_Float16)(acc[r] + bn);
    *(v8h*)(C + ((size_t)bb * N + ncol) * Sper + s) = o;   // contiguous in s
  }
}

// ---------------------------------------------------------------------------
// Flash attention with post-softmax dropout.
//  Qh,Kh: [B][S][D] f16 row-major.  Vt: [B][D][S] f16.  U: [B][S][S] f32.
//  out:   [B][S][D] f32.
// Block = 256 thr = 8 waves, one 16-row q-tile; wave w owns output d-chunk
// [128w,128w+128). Scores split across waves over D (no WMMA redundancy),
// reduced in LDS; softmax stats kept per-row in LDS; masked P shared in LDS.
// ---------------------------------------------------------------------------
__global__ __launch_bounds__(256) void flash_attn_f16(
    const _Float16* __restrict__ Qh, const _Float16* __restrict__ Kh,
    const _Float16* __restrict__ Vt, const float* __restrict__ U,
    float* __restrict__ out, int S, int D) {
  __shared__ __align__(16) _Float16 qs[16 * 1024];   // 32 KB  q-tile
  __shared__ float red[8][2][256];                   // 16 KB  per-wave partial scores
  __shared__ float ssum[2][256];                     //  2 KB  reduced scores (2 key tiles)
  __shared__ __align__(16) _Float16 Pls[16 * 32];    //  1 KB  masked exp(P)
  __shared__ float m_s[16], l_s[16], alpha_s[16];

  const int tid  = threadIdx.x;
  const int wave = tid >> 5;
  const int lane = tid & 31;
  const int b  = blockIdx.x;
  const int q0 = blockIdx.y * 16;
  const int half8 = (lane & 16) ? 8 : 0;
  const int koff  = (lane & 16) ? 16 : 0;
  const int dbase = wave * 128;

  { // stage q-tile into LDS (coalesced 16B copies)
    const uint4* src = (const uint4*)(Qh + ((size_t)b * S + q0) * D);
    uint4* dst = (uint4*)qs;
    const int nvec = (16 * 1024) / 8;
    for (int i = tid; i < nvec; i += 256) dst[i] = src[i];
  }
  if (tid < 16) { m_s[tid] = -INFINITY; l_s[tid] = 0.0f; }
  __syncthreads();

  v8f O[8] = {};

  const _Float16* Kb = Kh + (size_t)b * S * D;
  const _Float16* Vb = Vt + (size_t)b * D * S;
  const float*    Ub = U + ((size_t)b * S + q0) * S;

  for (int kv = 0; kv < S; kv += 32) {
    // ---- partial scores for this wave's d-range [dbase, dbase+128) ----
    v8f c0 = {}, c1 = {};
    const _Float16* krow = Kb + (size_t)(kv + (lane & 15)) * D;
    #pragma unroll
    for (int ds = 0; ds < 128; ds += 32) {
      const int d0 = dbase + ds;
      V16U a, bm;
      const _Float16* ap = qs + (lane & 15) * 1024 + d0;
      a.h[0] = *(const v8h*)(ap + koff);
      a.h[1] = *(const v8h*)(ap + koff + 8);
      bm.h[0] = *(const v8h*)(krow + d0 + koff);            // keys kv..kv+15
      bm.h[1] = *(const v8h*)(krow + d0 + koff + 8);
      c0 = __builtin_amdgcn_wmma_f32_16x16x32_f16(false, a.v, false, bm.v,
                                                  (short)0, c0, false, false);
      bm.h[0] = *(const v8h*)(krow + 16 * D + d0 + koff);   // keys kv+16..kv+31
      bm.h[1] = *(const v8h*)(krow + 16 * D + d0 + koff + 8);
      c1 = __builtin_amdgcn_wmma_f32_16x16x32_f16(false, a.v, false, bm.v,
                                                  (short)0, c1, false, false);
    }
    #pragma unroll
    for (int r = 0; r < 8; ++r) {
      const int e = (r + half8) * 16 + (lane & 15);  // row*16 + col
      red[wave][0][e] = c0[r];
      red[wave][1][e] = c1[r];
    }
    __syncthreads();

    // ---- cross-wave reduction: thread tid owns score element tid ----
    {
      float s0 = 0.f, s1 = 0.f;
      #pragma unroll
      for (int w = 0; w < 8; ++w) { s0 += red[w][0][tid]; s1 += red[w][1][tid]; }
      ssum[0][tid] = s0; ssum[1][tid] = s1;
    }
    __syncthreads();

    // ---- online softmax + dropout mask -> P (threads 0..15, one per q-row) ----
    if (tid < 16) {
      const int row = tid;
      float rmax = m_s[row];
      #pragma unroll
      for (int c = 0; c < 16; ++c) {
        rmax = fmaxf(rmax, ssum[0][row * 16 + c]);
        rmax = fmaxf(rmax, ssum[1][row * 16 + c]);
      }
      const float alpha = __expf(m_s[row] - rmax);
      float lsum = 0.f;
      const float* urow = Ub + (size_t)row * S + kv;
      for (int c = 0; c < 32; ++c) {
        const float sc = ssum[c >> 4][row * 16 + (c & 15)];
        const float p = __expf(sc - rmax);
        lsum += p;                                   // denominator is UNmasked
        Pls[row * 32 + c] = (_Float16)((urow[c] >= 0.2f) ? p * 1.25f : 0.0f);
      }
      m_s[row] = rmax;
      l_s[row] = l_s[row] * alpha + lsum;
      alpha_s[row] = alpha;
    }
    __syncthreads();

    // ---- O = O*alpha + P @ V  (8 WMMAs over this wave's 128-wide d-chunk) ----
    V16U pa;
    const _Float16* pp = Pls + (lane & 15) * 32;
    pa.h[0] = *(const v8h*)(pp + koff);
    pa.h[1] = *(const v8h*)(pp + koff + 8);
    float alv[8];
    #pragma unroll
    for (int r = 0; r < 8; ++r) alv[r] = alpha_s[r + half8];
    #pragma unroll
    for (int t = 0; t < 8; ++t) {
      #pragma unroll
      for (int r = 0; r < 8; ++r) O[t][r] *= alv[r];
      const _Float16* vp = Vb + (size_t)(dbase + t * 16 + (lane & 15)) * S + kv + koff;
      V16U bv;
      bv.h[0] = *(const v8h*)vp;
      bv.h[1] = *(const v8h*)(vp + 8);
      O[t] = __builtin_amdgcn_wmma_f32_16x16x32_f16(false, pa.v, false, bv.v,
                                                    (short)0, O[t], false, false);
    }
    if (kv + 32 < S) {  // pull next K/V tiles toward the WGP (global_prefetch_b8)
      __builtin_prefetch(krow + 32 * D + dbase, 0, 0);
      __builtin_prefetch(Vb + (size_t)(dbase + (lane & 15)) * S + kv + 32, 0, 0);
    }
    __syncthreads();
  }

  // ---- normalize by softmax denominator and store fp32 ----
  float linv[8];
  #pragma unroll
  for (int r = 0; r < 8; ++r) linv[r] = 1.0f / l_s[r + half8];
  float* orow = out + ((size_t)b * S + q0) * D;
  #pragma unroll
  for (int t = 0; t < 8; ++t) {
    const int dcol = dbase + t * 16 + (lane & 15);
    #pragma unroll
    for (int r = 0; r < 8; ++r)
      orow[(size_t)(r + half8) * D + dcol] = O[t][r] * linv[r];
  }
}

// ---------------------------------------------------------------------------
extern "C" void kernel_launch(void* const* d_in, const int* in_sizes, int n_in,
                              void* d_out, int out_size, void* d_ws, size_t ws_size,
                              hipStream_t stream) {
  (void)in_sizes; (void)n_in; (void)out_size; (void)ws_size;
  constexpr int B = 2, S = 4096, D = 1024;

  const float* x  = (const float*)d_in[0];
  const float* Wq = (const float*)d_in[1];
  const float* bq = (const float*)d_in[2];
  const float* Wk = (const float*)d_in[3];
  const float* bk = (const float*)d_in[4];
  const float* Wv = (const float*)d_in[5];
  const float* bv = (const float*)d_in[6];
  const float* du = (const float*)d_in[7];
  float* out = (float*)d_out;

  // workspace layout (all f16): xh | Wqh | Wkh | Wvh | Qh | Kh | Vt  (~70 MB)
  _Float16* xh  = (_Float16*)d_ws;
  _Float16* Wqh = xh  + (size_t)B * S * D;
  _Float16* Wkh = Wqh + (size_t)D * D;
  _Float16* Wvh = Wkh + (size_t)D * D;
  _Float16* Qh  = Wvh + (size_t)D * D;
  _Float16* Kh  = Qh  + (size_t)B * S * D;
  _Float16* Vt  = Kh  + (size_t)B * S * D;

  {
    int n4 = (B * S * D) / 4;
    cast_f32_to_f16_vec4<<<(n4 + 255) / 256, 256, 0, stream>>>(x, xh, n4);
    int w4 = (D * D) / 4;
    cast_f32_to_f16_vec4<<<(w4 + 255) / 256, 256, 0, stream>>>(Wq, Wqh, w4);
    cast_f32_to_f16_vec4<<<(w4 + 255) / 256, 256, 0, stream>>>(Wk, Wkh, w4);
    cast_f32_to_f16_vec4<<<(w4 + 255) / 256, 256, 0, stream>>>(Wv, Wvh, w4);
  }

  dim3 ggrid((B * S) / 16, D / 64);
  gemm_nt_bias_f16<<<ggrid, 128, 0, stream>>>(xh, Wqh, bq, Qh, B * S, D, D, 0, S);
  gemm_nt_bias_f16<<<ggrid, 128, 0, stream>>>(xh, Wkh, bk, Kh, B * S, D, D, 0, S);
  gemm_nt_bias_f16<<<ggrid, 128, 0, stream>>>(xh, Wvh, bv, Vt, B * S, D, D, 1, S);

  dim3 agrid(B, S / 16);
  flash_attn_f16<<<agrid, 256, 0, stream>>>(Qh, Kh, Vt, du, out, S, D);
}
